// FeatureRelationModule_27221502722563
// MI455X (gfx1250) — matrix operand order repaired and checked by reading, not compiled, not run on hardware
//
#include <hip/hip_runtime.h>
#include <hip/hip_bf16.h>
#include <math.h>

// CDNA5 WMMA vector types
typedef __attribute__((ext_vector_type(16))) __bf16 v16bf;
typedef __attribute__((ext_vector_type(8)))  __bf16 v8bf;
typedef __attribute__((ext_vector_type(4)))  __bf16 v4bf;
typedef __attribute__((ext_vector_type(8)))  float  v8f;

#define N_FEAT 512
#define FDIM   768
#define HDIM   256
#define NPAIRS (N_FEAT * (N_FEAT - 1) / 2)   // 130816

// Workspace layout (bytes):
//   Ebf : [512][768]  bf16               @ 0         (786,432 B)
//   Wt  : [2][256][768] bf16 (col-major) @ 786,432   (786,432 B)
//   H   : [2][512][256] f32              @ 1,572,864 (1,048,576 B)
#define WS_EBF_OFF 0
#define WS_WT_OFF  (N_FEAT * FDIM * 2)
#define WS_H_OFF   (WS_WT_OFF + 2 * HDIM * FDIM * 2)

// ---------------------------------------------------------------------------
// Prep 1: E f32 -> bf16, row-major (coalesced float4 in, v4bf out)
// ---------------------------------------------------------------------------
__global__ __launch_bounds__(256)
void cvt_E(const float* __restrict__ E, __bf16* __restrict__ Ebf) {
    const int idx = (blockIdx.x * blockDim.x + threadIdx.x) * 4;
    if (idx >= N_FEAT * FDIM) return;
    const float4 v = *(const float4*)(E + idx);
    v4bf o;
    o[0] = (__bf16)v.x; o[1] = (__bf16)v.y; o[2] = (__bf16)v.z; o[3] = (__bf16)v.w;
    *(v4bf*)(Ebf + idx) = o;
}

// ---------------------------------------------------------------------------
// Prep 2: W1 [1536][256] f32 -> Wt[half][col][k] bf16 (column-major per half)
// One-time strided gather; output writes are contiguous 8B per thread.
// ---------------------------------------------------------------------------
__global__ __launch_bounds__(256)
void cvt_W(const float* __restrict__ W1, __bf16* __restrict__ Wt) {
    const int gid = blockIdx.x * blockDim.x + threadIdx.x;   // 2*256*768/4 total
    if (gid >= 2 * HDIM * (FDIM / 4)) return;
    const int kc   = gid % (FDIM / 4);
    const int col  = (gid / (FDIM / 4)) % HDIM;
    const int half = gid / ((FDIM / 4) * HDIM);
    const int k    = kc * 4;
    const float* __restrict__ src = W1 + (half * FDIM + k) * HDIM + col;
    v4bf o;
    o[0] = (__bf16)src[0 * HDIM];
    o[1] = (__bf16)src[1 * HDIM];
    o[2] = (__bf16)src[2 * HDIM];
    o[3] = (__bf16)src[3 * HDIM];
    *(v4bf*)(Wt + (half * HDIM + col) * FDIM + k) = o;
}

// ---------------------------------------------------------------------------
// GEMM: H[half] = E @ W1[half].  One wave32 per 16x32 output tile (two 16x16
// WMMA accumulators sharing one A fragment).  All operands pre-converted bf16:
//   A frag: two contiguous v8bf (16B) loads per lane
//   B frag: one contiguous v16bf (32B) load per lane (column-major panel)
// VGPR layouts used here (contiguous k, same order for A and B):
//   A 16x32: lane<16 row=lane, K=k0..k0+15; lane>=16 K=k0+16..k0+31
//   B 32x16: lane<16 col=lane, K=k0..k0+15; lane>=16 col=lane-16, K=k0+16..31
// ---------------------------------------------------------------------------
__global__ __launch_bounds__(32)
void gemm_wmma(const __bf16* __restrict__ Ebf,   // [512][768]
               const __bf16* __restrict__ Wt,    // [2][256][768] col-major
               float* __restrict__ H)            // [2][512][256]
{
    const int tm   = blockIdx.x;            // row tile (0..31)
    const int tn2  = blockIdx.y;            // 32-col panel (0..7)
    const int half = blockIdx.z;
    const int lane = threadIdx.x;           // 0..31 (wave32)
    const bool hi  = lane >= 16;
    const int  lr  = lane & 15;

    const __bf16* __restrict__ Erow  = Ebf + (tm * 16 + lr) * FDIM;
    const __bf16* __restrict__ Wrow0 = Wt + (half * HDIM + tn2 * 32 + lr) * FDIM;
    const __bf16* __restrict__ Wrow1 = Wrow0 + 16 * FDIM;

    v8f c0 = {};
    v8f c1 = {};
    for (int k0 = 0; k0 < FDIM; k0 += 32) {
        if (k0 + 32 < FDIM) __builtin_prefetch(Erow + k0 + 32, 0, 1);

        const int ka = k0 + (hi ? 16 : 0);
        const int kb = k0 + (hi ? 16 : 0);

        const v8bf alo = *(const v8bf*)(Erow + ka);        // K = ka..ka+7
        const v8bf ahi = *(const v8bf*)(Erow + ka + 8);    // K = ka+8..ka+15
        const v16bf a = __builtin_shufflevector(alo, ahi,
            0, 1, 2, 3, 4, 5, 6, 7, 8, 9, 10, 11, 12, 13, 14, 15);

        const v16bf b0 = *(const v16bf*)(Wrow0 + kb);      // K = kb..kb+15
        const v16bf b1 = *(const v16bf*)(Wrow1 + kb);

        // (neg_a, A, neg_b, B, c_mod, C, reuse_a, reuse_b)
        c0 = __builtin_amdgcn_wmma_f32_16x16x32_bf16(false, a, false, b0,
                                                     (short)0, c0, false, false);
        c1 = __builtin_amdgcn_wmma_f32_16x16x32_bf16(false, a, false, b1,
                                                     (short)0, c1, false, false);
    }

    // 32-bit C/D 16x16: VGPR r -> M = r (+8 for lanes 16..31), N = lane&15
    float* __restrict__ O = H + half * N_FEAT * HDIM + tn2 * 32 + lr;
#pragma unroll
    for (int r = 0; r < 8; ++r) {
        const int m = r + (hi ? 8 : 0);
        O[(tm * 16 + m) * HDIM]      = c0[r];
        O[(tm * 16 + m) * HDIM + 16] = c1[r];
    }
}

// ---------------------------------------------------------------------------
// Pair scoring: one wave32 per pair p -> (i, j),
//   score = b2 + sum_k relu(A[i,k] + B[j,k] + b1[k]) * W2[k]
// ---------------------------------------------------------------------------
__device__ __forceinline__ int triu_base(int i) {
    return i * (2 * N_FEAT - i - 1) / 2;    // first pair index of row i
}

__global__ __launch_bounds__(256)
void pair_score(const float* __restrict__ H,   // [2][512][256]
                const float* __restrict__ b1,  // [256]
                const float* __restrict__ W2,  // [256]
                const float* __restrict__ b2,  // [1]
                int*   __restrict__ iout,      // [P]
                int*   __restrict__ jout,      // [P]
                float* __restrict__ sout)      // [P]
{
    const int gid  = blockIdx.x * blockDim.x + threadIdx.x;
    const int p    = gid >> 5;      // wave index == pair index
    const int lane = gid & 31;
    if (p >= NPAIRS) return;

    // Invert triu(k=1) row-major pair index: float seed + exact integer fixup.
    const float t = 2.0f * (float)N_FEAT - 1.0f;
    int i = (int)((t - sqrtf(t * t - 8.0f * (float)p)) * 0.5f);
    if (i < 0) i = 0;
    if (i > N_FEAT - 2) i = N_FEAT - 2;
    while (i < N_FEAT - 2 && triu_base(i + 1) <= p) ++i;
    while (i > 0 && triu_base(i) > p) --i;
    const int j = p - triu_base(i) + i + 1;

    const float* __restrict__ Ai = H + i * HDIM;                  // A half
    const float* __restrict__ Bj = H + N_FEAT * HDIM + j * HDIM;  // B half

    float s = 0.0f;
#pragma unroll
    for (int cchunk = 0; cchunk < 2; ++cchunk) {
        const int k = cchunk * 128 + lane * 4;
        const float4 av  = *(const float4*)(Ai + k);
        const float4 bv  = *(const float4*)(Bj + k);
        const float4 b1v = *(const float4*)(b1 + k);
        const float4 wv  = *(const float4*)(W2 + k);
        float h;
        h = av.x + bv.x + b1v.x; h = h > 0.0f ? h : 0.0f; s += h * wv.x;
        h = av.y + bv.y + b1v.y; h = h > 0.0f ? h : 0.0f; s += h * wv.y;
        h = av.z + bv.z + b1v.z; h = h > 0.0f ? h : 0.0f; s += h * wv.z;
        h = av.w + bv.w + b1v.w; h = h > 0.0f ? h : 0.0f; s += h * wv.w;
    }

    // wave32 tree reduction
#pragma unroll
    for (int m = 16; m >= 1; m >>= 1) s += __shfl_xor(s, m, 32);

    if (lane == 0) {
        iout[p] = i;
        jout[p] = j;
        sout[p] = s + b2[0];
    }
}

// ---------------------------------------------------------------------------
extern "C" void kernel_launch(void* const* d_in, const int* in_sizes, int n_in,
                              void* d_out, int out_size, void* d_ws, size_t ws_size,
                              hipStream_t stream) {
    (void)in_sizes; (void)n_in; (void)out_size; (void)ws_size;

    const float* E  = (const float*)d_in[0];   // [512*768]
    const float* W1 = (const float*)d_in[1];   // [1536*256]
    const float* b1 = (const float*)d_in[2];   // [256]
    const float* W2 = (const float*)d_in[3];   // [256]
    const float* b2 = (const float*)d_in[4];   // [1]

    char* ws = (char*)d_ws;
    __bf16* Ebf = (__bf16*)(ws + WS_EBF_OFF);
    __bf16* Wt  = (__bf16*)(ws + WS_WT_OFF);
    float*  H   = (float*)(ws + WS_H_OFF);

    // d_out: [P] int32 i_idx | [P] int32 j_idx | [P] f32 scores (flat concat)
    int*   iout = (int*)d_out;
    int*   jout = iout + NPAIRS;
    float* sout = (float*)d_out + 2 * NPAIRS;

    // Prep: one-time bf16 conversion / layout transforms
    cvt_E<<<(N_FEAT * FDIM / 4 + 255) / 256, 256, 0, stream>>>(E, Ebf);
    cvt_W<<<(2 * HDIM * (FDIM / 4) + 255) / 256, 256, 0, stream>>>(W1, Wt);

    // GEMM: 16x32 tile per wave32
    dim3 g1(N_FEAT / 16, HDIM / 32, 2);        // (32, 8, 2)
    gemm_wmma<<<g1, 32, 0, stream>>>(Ebf, Wt, H);

    const int threads = 256;                    // 8 waves per block
    const int blocks  = (NPAIRS * 32 + threads - 1) / threads;  // 16352
    pair_score<<<blocks, threads, 0, stream>>>(H, b1, W2, b2, iout, jout, sout);
}
